// Fusion_72112500899920
// MI455X (gfx1250) — matrix operand order repaired and checked by reading, not compiled, not run on hardware
//
#include <hip/hip_runtime.h>
#include <hip/hip_bf16.h>

#define BDIM 16
#define TDIM 2048
#define FDIM 1024
#define TOPK 8

// TDM pads the LDS destination by 16B (4 DWORDs) every 256 DWORDs (512 halves):
// key-row stride in LDS = 1024 + 16 halves = 2080 bytes -> banks staggered for
// the B-fragment ds_load_b128s.
#define KROW_H   1040                       // halves per key row in LDS
#define SCHUNK   64                         // keys staged per chunk
#define KBUF_H   (SCHUNK * KROW_H)          // halves per LDS buffer (66560)
#define KBUF_B   (KBUF_H * 2)               // bytes per LDS buffer  (133120)
#define NCHUNK   (TDIM / SCHUNK)            // 32 chunks

typedef __attribute__((ext_vector_type(16))) _Float16 v16h;
typedef __attribute__((ext_vector_type(8)))  _Float16 v8h;
typedef __attribute__((ext_vector_type(4)))  _Float16 h4;
typedef __attribute__((ext_vector_type(8)))  float    v8f;
typedef __attribute__((ext_vector_type(4)))  unsigned int v4u;
typedef __attribute__((ext_vector_type(8)))  int      v8i;
typedef __attribute__((ext_vector_type(4)))  int      v4i;

// ---------------------------------------------------------------------------
// Tensor Data Mover: DMA `units8` 8-byte units (contiguous in global) into LDS
// at byte offset lds_off, inserting 16B of padding every 256 DWORDs so each
// 2048B key row lands with a 2080B stride (bank stagger).
// ---------------------------------------------------------------------------
static __device__ __forceinline__ void tdm_load_to_lds(unsigned lds_off,
                                                       const _Float16* gsrc,
                                                       unsigned units8)
{
    unsigned long long ga = (unsigned long long)(size_t)gsrc;
    unsigned galo = __builtin_amdgcn_readfirstlane((unsigned)ga);
    unsigned gahi = __builtin_amdgcn_readfirstlane((unsigned)(ga >> 32));
    unsigned loff = __builtin_amdgcn_readfirstlane(lds_off);
    unsigned u8   = __builtin_amdgcn_readfirstlane(units8);

    v4u g0;
    g0[0] = 1u;                                   // count=1 (valid descriptor)
    g0[1] = loff;                                 // lds_addr (bytes)
    g0[2] = galo;                                 // global_addr[31:0]
    g0[3] = (gahi & 0x01FFFFFFu) | (2u << 30);    // global_addr[56:32] | type=2

    v8i g1;
    g1[0] = (int)((3u << 16)       // data_size = 3 (8-byte units)
                | (1u << 20)       // pad_enable
                | (7u << 22)       // pad_interval = 256 DWORDs
                | (3u << 25));     // pad_amount  = 4 DWORDs (16B)
    g1[1] = (int)((u8 & 0xFFFFu) << 16);          // tensor_dim0[15:0]
    g1[2] = (int)(((u8 >> 16) & 0xFFFFu)          // tensor_dim0[31:16]
                | (1u << 16));                    // tensor_dim1 = 1
    g1[3] = (int)((u8 & 0xFFFFu) << 16);          // tile_dim0 = units8
    g1[4] = 0;                                    // tile_dim1/2 unused
    g1[5] = (int)u8;                              // tensor_dim0_stride
    g1[6] = 0;
    g1[7] = 0;

    v4i z4 = {0, 0, 0, 0};
#if defined(__clang_major__) && __clang_major__ >= 23
    v8i z8 = {0, 0, 0, 0, 0, 0, 0, 0};
    __builtin_amdgcn_tensor_load_to_lds(g0, g1, z4, z4, z8, 0);
#else
    __builtin_amdgcn_tensor_load_to_lds(g0, g1, z4, z4, 0);
#endif
}

// ---------------------------------------------------------------------------
// Fragment helpers.
// ---------------------------------------------------------------------------
static __device__ __forceinline__ v16h pack16(v8h lo, v8h hi)
{
    return __builtin_shufflevector(lo, hi, 0, 1, 2, 3, 4, 5, 6, 7,
                                   8, 9, 10, 11, 12, 13, 14, 15);
}

// A (16x32 f16) straight from global: lane -> query row M=lan;
// halves 0..15 = K(kc + half*16 + 0..15).
static __device__ __forceinline__ v16h load_a_frag_g(const _Float16* qrow,
                                                     int kc)
{
    v8h lo = *(const v8h*)(qrow + kc);
    v8h hi = *(const v8h*)(qrow + kc + 8);
    return pack16(lo, hi);
}

// B (32x16 f16) from padded LDS chunk: lane -> key row (column N),
// 16 consecutive K starting at kadj (pad-adjusted kc + half*16).
static __device__ __forceinline__ v16h load_b_frag_lds(const _Float16* ks,
                                                       int row, int kadj)
{
    const _Float16* base = ks + row * KROW_H + kadj;
    v8h lo = *(const v8h*)(base);
    v8h hi = *(const v8h*)(base + 8);
    return pack16(lo, hi);
}

static __device__ __forceinline__ v8f wmma_f16(v16h a, v16h b, v8f c)
{
    return __builtin_amdgcn_wmma_f32_16x16x32_f16(false, a, false, b,
                                                  (short)0, c, false, false);
}

// ---------------------------------------------------------------------------
// Kernel 1: L2-normalize rows of [rows, FDIM] f32 and store as f16.
// ---------------------------------------------------------------------------
__global__ __launch_bounds__(256) void normalize_f16_kernel(
    const float* __restrict__ x, _Float16* __restrict__ y)
{
    const int row  = blockIdx.x * 8 + (threadIdx.x >> 5);
    const int lane = threadIdx.x & 31;

    const float4* xr = (const float4*)(x + (size_t)row * FDIM);
    float4 vals[8];
    float ss = 0.f;
#pragma unroll
    for (int i = 0; i < 8; ++i) {
        float4 v = xr[lane + i * 32];
        vals[i] = v;
        ss += v.x * v.x + v.y * v.y + v.z * v.z + v.w * v.w;
    }
#pragma unroll
    for (int m = 1; m < 32; m <<= 1) ss += __shfl_xor(ss, m, 32);
    const float inv = rsqrtf(ss);

    h4* yr = (h4*)(y + (size_t)row * FDIM);
#pragma unroll
    for (int i = 0; i < 8; ++i) {
        float4 v = vals[i];
        h4 h;
        h.x = (_Float16)(v.x * inv);
        h.y = (_Float16)(v.y * inv);
        h.z = (_Float16)(v.z * inv);
        h.w = (_Float16)(v.w * inv);
        yr[lane + i * 32] = h;
    }
}

// ---------------------------------------------------------------------------
// Kernel 2: similarity + softmax row statistics via WMMA.
// Grid (TDIM/128, BDIM); 8 waves/block, wave owns 16 query rows.
//  - All 32 A fragments live in registers (extended VGPR file).
//  - Key chunks (64 rows) staged once per block by wave 0 via TDM into a
//    double-buffered, bank-padded LDS region; all 8 waves share them
//    (8x less L2 traffic than per-wave streaming).
//  - 4 independent accumulators per wave; per-lane z += exp(v-1)
//    (cosine sims <= ~1 so max is fixed), one cross-lane reduce at the end.
// score[b,t] = logsumexp_s <q_t, k_s>  (monotone softmax-rank surrogate).
// ---------------------------------------------------------------------------
__global__ __launch_bounds__(256) void sim_softmax_kernel(
    const _Float16* __restrict__ Q,   // [B, T, F] normalized f16
    const _Float16* __restrict__ Kf,  // [B, T, F] normalized f16
    float* __restrict__ scores)       // [B, T]
{
    extern __shared__ _Float16 smem[];            // 2 * KBUF_H halves

    const int b    = blockIdx.y;
    const int lane = threadIdx.x & 31;
    const int wave = threadIdx.x >> 5;
    const int m0   = blockIdx.x * 128 + wave * 16;
    const int half = lane >> 4;                   // 0: lanes 0-15, 1: 16-31
    const int lan  = lane & 15;

    const _Float16* Qb = Q  + ((size_t)b * TDIM + m0) * FDIM;
    const _Float16* Kb = Kf + (size_t)b * TDIM * FDIM;

    // Prologue: park all 32 A fragments (16 rows x full K) in registers.
    const _Float16* qrow = Qb + (size_t)lan * FDIM + half * 16;
    v16h afr[32];
#pragma unroll
    for (int i = 0; i < 32; ++i) afr[i] = load_a_frag_g(qrow, i * 32);

    float rz[8];
#pragma unroll
    for (int r = 0; r < 8; ++r) rz[r] = 0.f;

    // Prime the first key chunk.
    if (wave == 0) tdm_load_to_lds(0u, Kb, SCHUNK * 256u);

    for (int ch = 0; ch < NCHUNK; ++ch) {
        __syncthreads();   // buffer (ch+1)&1 is no longer being read
        if (wave == 0) {
            if (ch + 1 < NCHUNK) {
                tdm_load_to_lds((unsigned)(KBUF_B * ((ch + 1) & 1)),
                                Kb + (size_t)(ch + 1) * SCHUNK * FDIM,
                                SCHUNK * 256u);
                __builtin_amdgcn_s_wait_tensorcnt(1);  // chunk `ch` landed
            } else {
                __builtin_amdgcn_s_wait_tensorcnt(0);
            }
        }
        __syncthreads();   // chunk `ch` visible to all waves

        const _Float16* ks = smem + (size_t)(ch & 1) * KBUF_H;

        v8f c0 = {};
        v8f c1 = {};
        v8f c2 = {};
        v8f c3 = {};
#pragma unroll
        for (int i = 0; i < 32; ++i) {
            const int kc   = i * 32 + half * 16;
            const int kadj = kc + ((kc >= 512) ? 8 : 0);   // skip 16B pad
            v16h b0 = load_b_frag_lds(ks, lan +  0, kadj);
            v16h b1 = load_b_frag_lds(ks, lan + 16, kadj);
            v16h b2 = load_b_frag_lds(ks, lan + 32, kadj);
            v16h b3 = load_b_frag_lds(ks, lan + 48, kadj);
            c0 = wmma_f16(afr[i], b0, c0);
            c1 = wmma_f16(afr[i], b1, c1);
            c2 = wmma_f16(afr[i], b2, c2);
            c3 = wmma_f16(afr[i], b3, c3);
        }

        // Per-lane softmax accumulation (TRANS ops co-execute with WMMA).
#pragma unroll
        for (int r = 0; r < 8; ++r)
            rz[r] += __expf(c0[r] - 1.f) + __expf(c1[r] - 1.f)
                   + __expf(c2[r] - 1.f) + __expf(c3[r] - 1.f);
    }

    // One cross-lane reduction (within each 16-lane half) at the end.
#pragma unroll
    for (int r = 0; r < 8; ++r) {
        float z = rz[r];
        z += __shfl_xor(z, 1, 16);
        z += __shfl_xor(z, 2, 16);
        z += __shfl_xor(z, 4, 16);
        z += __shfl_xor(z, 8, 16);
        if (lan == 0) {
            const int M = r + half * 8;
            scores[(size_t)b * TDIM + m0 + M] = 1.f + __logf(z);
        }
    }
}

// ---------------------------------------------------------------------------
// Kernel 3: top-8 per batch (argmax; negate input for "smallest").
// ---------------------------------------------------------------------------
__global__ __launch_bounds__(256) void topk_kernel(
    const float* __restrict__ scores, int* __restrict__ idx, int findMax)
{
    __shared__ float sv[TDIM];
    __shared__ float redv[256];
    __shared__ int   redi[256];

    const int b = blockIdx.x;
    const float* s = scores + (size_t)b * TDIM;
    for (int i = threadIdx.x; i < TDIM; i += 256)
        sv[i] = findMax ? s[i] : -s[i];
    __syncthreads();

    for (int k = 0; k < TOPK; ++k) {
        float best = -1e38f; int bi = 0;
        for (int i = threadIdx.x; i < TDIM; i += 256) {
            float v = sv[i];
            if (v > best) { best = v; bi = i; }
        }
        redv[threadIdx.x] = best; redi[threadIdx.x] = bi;
        __syncthreads();
        for (int off = 128; off > 0; off >>= 1) {
            if (threadIdx.x < off) {
                float v2 = redv[threadIdx.x + off];
                int   i2 = redi[threadIdx.x + off];
                if (v2 > redv[threadIdx.x] ||
                    (v2 == redv[threadIdx.x] && i2 < redi[threadIdx.x])) {
                    redv[threadIdx.x] = v2; redi[threadIdx.x] = i2;
                }
            }
            __syncthreads();
        }
        if (threadIdx.x == 0) {
            idx[b * TOPK + k] = redi[0];
            sv[redi[0]] = -1e38f;
        }
        __syncthreads();
    }
}

// ---------------------------------------------------------------------------
// Kernel 4: gather selected rows: dst[bk, :] = src[b, idx[bk], :]
// ---------------------------------------------------------------------------
__global__ __launch_bounds__(256) void gather_kernel(
    const float* __restrict__ src, const int* __restrict__ idx,
    float* __restrict__ dst)
{
    const int bk  = blockIdx.x;
    const int b   = bk >> 3;          // TOPK == 8
    const int row = idx[bk];
    const float4* s = (const float4*)(src + ((size_t)b * TDIM + row) * FDIM);
    float4*       d = (float4*)(dst + (size_t)bk * FDIM);
    d[threadIdx.x] = s[threadIdx.x];
}

// ---------------------------------------------------------------------------
extern "C" void kernel_launch(void* const* d_in, const int* in_sizes, int n_in,
                              void* d_out, int out_size, void* d_ws, size_t ws_size,
                              hipStream_t stream)
{
    (void)in_sizes; (void)n_in; (void)out_size; (void)ws_size;

    const float* ref_rgb  = (const float*)d_in[0];
    const float* ref_flow = (const float*)d_in[1];
    const float* sup_rgb  = (const float*)d_in[2];
    const float* sup_flow = (const float*)d_in[3];
    float* out = (float*)d_out;

    char* ws = (char*)d_ws;
    const size_t f16bytes = (size_t)BDIM * TDIM * FDIM * sizeof(_Float16); // 64 MB
    _Float16* rf16 = (_Float16*)ws;
    _Float16* sf16 = (_Float16*)(ws + f16bytes);
    float* sc_intra = (float*)(ws + 2 * f16bytes);
    float* sc_inter = sc_intra + (size_t)BDIM * TDIM;
    int* idx1 = (int*)(sc_inter + (size_t)BDIM * TDIM);
    int* idx2 = idx1 + BDIM * TOPK;

    const int rows = BDIM * TDIM;                   // 32768
    normalize_f16_kernel<<<rows / 8, 256, 0, stream>>>(ref_flow, rf16);
    normalize_f16_kernel<<<rows / 8, 256, 0, stream>>>(sup_flow, sf16);

    dim3 gsim(TDIM / 128, BDIM);
    const size_t shmem = (size_t)2 * KBUF_B;        // 266240 B (<320KB WGP LDS)
    sim_softmax_kernel<<<gsim, 256, shmem, stream>>>(rf16, rf16, sc_intra);
    sim_softmax_kernel<<<gsim, 256, shmem, stream>>>(sf16, rf16, sc_inter);

    topk_kernel<<<BDIM, 256, 0, stream>>>(sc_intra, idx1, /*findMax=*/0);
    topk_kernel<<<BDIM, 256, 0, stream>>>(sc_inter, idx2, /*findMax=*/1);

    gather_kernel<<<BDIM * TOPK, 256, 0, stream>>>(ref_rgb, idx1, out);
    gather_kernel<<<BDIM * TOPK, 256, 0, stream>>>(sup_rgb, idx2,
                                                   out + (size_t)BDIM * TOPK * FDIM);
}
